// DCRNNEncoder_75694503624817
// MI455X (gfx1250) — matrix operand order repaired and checked by reading, not compiled, not run on hardware
//
#include <hip/hip_runtime.h>
#include <math.h>

// ---------------------------------------------------------------------------
// DCRNN encoder for MI455X (gfx1250, wave32, WMMA + async LDS staging).
//   N=1024 nodes, H=64 hidden, B=8 batch, T=12 steps, L=2 layers,
//   K=2 diffusion steps, 2 supports -> M=5 diffusion mats.
// GEMMs: f16 data, f32 accumulation via v_wmma_f32_16x16x32_f16.
// A-tiles staged into LDS once per block via global_load_async_to_lds_b128.
// Whole working set (~55MB) is L2-resident on MI455X (192MB L2).
// ---------------------------------------------------------------------------

typedef _Float16 h16;
typedef __attribute__((ext_vector_type(8)))  _Float16 v8h;
typedef __attribute__((ext_vector_type(16))) _Float16 v16h;
typedef __attribute__((ext_vector_type(8)))  float    v8f;

#define N_NODES 1024
#define HDIM    64
#define BATCH   8
#define TSTEPS  12
#define NNZE    16384
#define MDIFF   5
#define NH      (N_NODES * HDIM)   // 65536
#define BNH     (BATCH * NH)       // 524288
#define BN      (BATCH * N_NODES)  // 8192 GEMM rows
#define ROWS_PB 32                 // A rows staged per block (2 sub-tiles/wave)
#define KP_MAX  640

// ---------------------------------------------------------------------------
// Build x_cat [N, F*B] (col = f*B + b) from xi [B, N*in_f] and h [B, N*H]
// (optionally r*h), and write the m=0 slot of the f16 GEMM A-matrix
// X[b*N+n, f*M + 0] (row stride Kp).
// ---------------------------------------------------------------------------
__global__ void concat_build(const float* __restrict__ xi, int in_f,
                             const float* __restrict__ h,
                             const float* __restrict__ r, int use_r,
                             float* __restrict__ xcat,
                             h16* __restrict__ X, int Kp)
{
    const int F  = in_f + HDIM;
    const int FB = F * BATCH;
    const int total = N_NODES * FB;
    for (int i = blockIdx.x * blockDim.x + threadIdx.x; i < total;
         i += gridDim.x * blockDim.x) {
        int n   = i / FB;
        int rem = i - n * FB;
        int f   = rem / BATCH;
        int b   = rem - f * BATCH;
        float v;
        if (f < in_f) {
            v = xi[(size_t)(b * N_NODES + n) * in_f + f];
        } else {
            int hidx = b * NH + n * HDIM + (f - in_f);
            v = h[hidx];
            if (use_r) v *= r[hidx];
        }
        xcat[i] = v;
        X[(size_t)(b * N_NODES + n) * Kp + f * MDIFF] = (h16)v;
    }
}

// ---------------------------------------------------------------------------
// COO SpMM: acc[row,:] += val * xin[col,:].  One block per nonzero,
// coalesced f32 atomics across FB columns (L2-resident, low contention).
// ---------------------------------------------------------------------------
__global__ void spmm_atomic(const int* __restrict__ rows,
                            const int* __restrict__ cols,
                            const float* __restrict__ vals,
                            const float* __restrict__ xin,
                            float* __restrict__ acc, int FB)
{
    const int e = blockIdx.x;
    const int rr = rows[e];
    const int cc = cols[e];
    const float v = vals[e];
    const float* __restrict__ src = xin + (size_t)cc * FB;
    float* __restrict__ dst       = acc + (size_t)rr * FB;
    for (int c = threadIdx.x; c < FB; c += blockDim.x)
        atomicAdd(&dst[c], v * src[c]);
}

// ---------------------------------------------------------------------------
// Chebyshev combine + scatter into f16 A-matrix slot m:
//   xk = alpha*acc + beta*xm2 ; X[b*N+n, f*M + m] = (f16)xk
// ---------------------------------------------------------------------------
__global__ void cheb_writeX(const float* __restrict__ acc,
                            const float* __restrict__ xm2,
                            float alpha, float beta,
                            h16* __restrict__ X, int Kp, int F, int m)
{
    const int FB = F * BATCH;
    const int total = N_NODES * FB;
    for (int i = blockIdx.x * blockDim.x + threadIdx.x; i < total;
         i += gridDim.x * blockDim.x) {
        int n   = i / FB;
        int rem = i - n * FB;
        int f   = rem / BATCH;
        int b   = rem - f * BATCH;
        float v = alpha * acc[i] + beta * xm2[i];
        X[(size_t)(b * N_NODES + n) * Kp + f * MDIFF + m] = (h16)v;
    }
}

// ---------------------------------------------------------------------------
// Pack f32 weights [K, Ncols] into WMMA-B-fragment order f16:
//   Wp[(n*KT + kt)*32 + kk] = (k = kt*32+kk) < K ? W[k*Ncols + n] : 0
// so a lane's B fragment is two contiguous 16B loads.
// ---------------------------------------------------------------------------
__global__ void pack_weights(const float* __restrict__ W, int K, int Ncols,
                             int KT, h16* __restrict__ Wp)
{
    const int total = Ncols * KT * 32;
    for (int i = blockIdx.x * blockDim.x + threadIdx.x; i < total;
         i += gridDim.x * blockDim.x) {
        int n   = i / (KT * 32);
        int rem = i - n * (KT * 32);
        int kt  = rem >> 5;
        int kk  = rem & 31;
        int k   = kt * 32 + kk;
        Wp[i] = (k < K) ? (h16)W[(size_t)k * Ncols + n] : (h16)0.0f;
    }
}

// ---------------------------------------------------------------------------
// WMMA fragment helpers (wave32; C per CDNA5 ISA 7.12.2, A/B in this code's contiguous k order).
// A 16x32 f16: lane m=lane&15; halves = K{0..15} (lanes 0-15)
//                              or      K{16..31} (lanes 16-31).
// B 32x16 f16: lane n=lane&15; half j = K{j} (lanes 0-15) / K{16+j}.
// C 16x16 f32: VGPR i -> row i (lanes 0-15) / 8+i ; col = lane&15.
// ---------------------------------------------------------------------------
__device__ __forceinline__ v16h cat16(v8h lo, v8h hi)
{
    return __builtin_shufflevector(lo, hi, 0, 1, 2, 3, 4, 5, 6, 7,
                                   8, 9, 10, 11, 12, 13, 14, 15);
}

// Stage `bytes` of the contiguous A tile at gsrc into LDS at shA using
// gfx1250 async copy (ASYNCcnt path), then sync the block.
#define STAGE_A_ASYNC(shA, gsrc, bytes)                                        \
    {                                                                          \
        uint32_t shBase_ = (uint32_t)(uintptr_t)(shA);                         \
        const char* gsrc_ = (const char*)(gsrc);                               \
        const int nChunks_ = (int)((bytes) >> 4);                              \
        for (int ch_ = threadIdx.x; ch_ < nChunks_; ch_ += blockDim.x) {       \
            uint32_t lds_ = shBase_ + (uint32_t)(ch_ << 4);                    \
            uint64_t ga_  = (uint64_t)(uintptr_t)(gsrc_ + ((size_t)ch_ << 4)); \
            asm volatile("global_load_async_to_lds_b128 %0, %1, off"           \
                         :: "v"(lds_), "v"(ga_) : "memory");                   \
        }                                                                      \
        asm volatile("s_wait_asynccnt 0" ::: "memory");                        \
        __syncthreads();                                                       \
    }

// Main K loop: one B fragment, two A fragments from LDS, two WMMAs per tile.
#define WMMA_MAIN_LOOP(shA, Kp, KT, Bfrag, c0, c1, lm, laneHi)                 \
    for (int kt = 0; kt < (KT); ++kt) {                                        \
        __builtin_prefetch((Bfrag) + (kt + 1) * 32, 0, 0);                     \
        v8h blo = *(const v8h*)((Bfrag) + kt * 32);                            \
        v8h bhi = *(const v8h*)((Bfrag) + kt * 32 + 8);                        \
        v16h bfr = cat16(blo, bhi);                                            \
        const int aoff = kt * 32 + ((laneHi) << 4);                            \
        v8h a0lo = *(const v8h*)&(shA)[(lm) * (Kp) + aoff];                    \
        v8h a0hi = *(const v8h*)&(shA)[(lm) * (Kp) + aoff + 8];                \
        v8h a1lo = *(const v8h*)&(shA)[((lm) + 16) * (Kp) + aoff];             \
        v8h a1hi = *(const v8h*)&(shA)[((lm) + 16) * (Kp) + aoff + 8];         \
        c0 = __builtin_amdgcn_wmma_f32_16x16x32_f16(                           \
                 false, cat16(a0lo, a0hi), false, bfr, (short)0, c0,           \
                 false, false);                                                \
        c1 = __builtin_amdgcn_wmma_f32_16x16x32_f16(                           \
                 false, cat16(a1lo, a1hi), false, bfr, (short)0, c1,           \
                 false, false);                                                \
    }

// ---------------------------------------------------------------------------
// Gates GEMM: sigmoid(X[8192,Kp] @ Wg[Kp,128] + bg) -> r [B,N*H], u [B,N*H].
// Block = 256 threads = 8 waves = 8 column tiles; 32 rows staged in LDS,
// each wave computes two stacked 16x16 C tiles. grid.x = 8192/32.
// ---------------------------------------------------------------------------
__global__ void wmma_gates(const h16* __restrict__ X, int Kp, int KT,
                           const h16* __restrict__ Wp,
                           const float* __restrict__ bias,
                           float* __restrict__ rbuf, float* __restrict__ ubuf)
{
    __shared__ h16 shA[ROWS_PB * KP_MAX];

    const int lane   = threadIdx.x & 31;
    const int wave   = threadIdx.x >> 5;
    const int laneHi = lane >> 4;
    const int lm     = lane & 15;
    const int rowBase = blockIdx.x * ROWS_PB;
    const int colBase = wave * 16;

    STAGE_A_ASYNC(shA, X + (size_t)rowBase * Kp, (size_t)ROWS_PB * Kp * 2);

    const h16* Bfrag = Wp + ((size_t)(colBase + lm) * KT) * 32 + (laneHi << 4);
    v8f c0 = {}, c1 = {};
    WMMA_MAIN_LOOP(shA, Kp, KT, Bfrag, c0, c1, lm, laneHi);

    const int col = colBase + lm;
    const float bv = bias[col];
#pragma unroll
    for (int st = 0; st < 2; ++st) {
        const v8f& c = st ? c1 : c0;
#pragma unroll
        for (int i = 0; i < 8; ++i) {
            int row = rowBase + st * 16 + i + laneHi * 8;
            int b = row >> 10;          // / N
            int n = row & 1023;         // % N
            float g = 1.0f / (1.0f + __expf(-(c[i] + bv)));
            if (col < HDIM) rbuf[b * NH + n * HDIM + col] = g;
            else            ubuf[b * NH + n * HDIM + (col - HDIM)] = g;
        }
    }
}

// ---------------------------------------------------------------------------
// Candidate GEMM + GRU update:
//   cand = tanh(X @ Wc + bc); h_new = u*h + (1-u)*cand
// Block = 128 threads = 4 waves = 4 column tiles (64 cols); 32 rows in LDS.
// ---------------------------------------------------------------------------
__global__ void wmma_cand_update(const h16* __restrict__ X, int Kp, int KT,
                                 const h16* __restrict__ Wp,
                                 const float* __restrict__ bias,
                                 const float* __restrict__ ubuf,
                                 float* __restrict__ hbuf,
                                 float* __restrict__ outs_t,
                                 float* __restrict__ lasth, int write_last)
{
    __shared__ h16 shA[ROWS_PB * KP_MAX];

    const int lane   = threadIdx.x & 31;
    const int wave   = threadIdx.x >> 5;
    const int laneHi = lane >> 4;
    const int lm     = lane & 15;
    const int rowBase = blockIdx.x * ROWS_PB;
    const int colBase = wave * 16;

    STAGE_A_ASYNC(shA, X + (size_t)rowBase * Kp, (size_t)ROWS_PB * Kp * 2);

    const h16* Bfrag = Wp + ((size_t)(colBase + lm) * KT) * 32 + (laneHi << 4);
    v8f c0 = {}, c1 = {};
    WMMA_MAIN_LOOP(shA, Kp, KT, Bfrag, c0, c1, lm, laneHi);

    const int col = colBase + lm;
    const float bv = bias[col];
#pragma unroll
    for (int st = 0; st < 2; ++st) {
        const v8f& c = st ? c1 : c0;
#pragma unroll
        for (int i = 0; i < 8; ++i) {
            int row = rowBase + st * 16 + i + laneHi * 8;
            int b = row >> 10;
            int n = row & 1023;
            int idx = b * NH + n * HDIM + col;
            float cand = tanhf(c[i] + bv);
            float uv = ubuf[idx];
            float hn = uv * hbuf[idx] + (1.0f - uv) * cand;
            hbuf[idx]   = hn;
            outs_t[idx] = hn;
            if (write_last) lasth[idx] = hn;
        }
    }
}

// ---------------------------------------------------------------------------
// Host-side orchestration (graph-capture safe: kernels + MemsetAsync/
// MemcpyAsync on `stream` only).
// ---------------------------------------------------------------------------
extern "C" void kernel_launch(void* const* d_in, const int* in_sizes, int n_in,
                              void* d_out, int out_size, void* d_ws,
                              size_t ws_size, hipStream_t stream)
{
    (void)in_sizes; (void)n_in; (void)out_size; (void)ws_size;

    const float* inputs = (const float*)d_in[0];   // [T, B, N*2]
    const float* init_h = (const float*)d_in[1];   // [L, B, N*H]
    const int*   srows  = (const int*)  d_in[2];   // [2, NNZ]
    const int*   scols  = (const int*)  d_in[3];
    const float* svals  = (const float*)d_in[4];
    const float* Wg[2]  = { (const float*)d_in[5], (const float*)d_in[9]  };
    const float* bg[2]  = { (const float*)d_in[6], (const float*)d_in[10] };
    const float* Wc[2]  = { (const float*)d_in[7], (const float*)d_in[11] };
    const float* bc[2]  = { (const float*)d_in[8], (const float*)d_in[12] };

    float* out      = (float*)d_out;
    float* hid_out  = out;             // [L, B, N*H]
    float* cur_out  = out + 2 * BNH;   // [T, B, N*H]

    // ---- workspace carve-up (256B aligned) --------------------------------
    char* w = (char*)d_ws;
    auto carve = [&](size_t bytes) -> void* {
        void* p = (void*)w;
        w += (bytes + 255) & ~(size_t)255;
        return p;
    };
    const int Kp[2] = { 352, 640 };        // padded K (F*M=330 -> 352; 640)
    h16*   X     = (h16*)  carve((size_t)BN * 640 * sizeof(h16));
    float* xcat  = (float*)carve((size_t)N_NODES * 1024 * sizeof(float));
    float* acc1  = (float*)carve((size_t)N_NODES * 1024 * sizeof(float));
    float* acc2  = (float*)carve((size_t)N_NODES * 1024 * sizeof(float));
    h16*   Wgp[2]; h16* Wcp[2];
    Wgp[0] = (h16*)carve((size_t)128 * Kp[0] * sizeof(h16));
    Wcp[0] = (h16*)carve((size_t)64  * Kp[0] * sizeof(h16));
    Wgp[1] = (h16*)carve((size_t)128 * Kp[1] * sizeof(h16));
    Wcp[1] = (h16*)carve((size_t)64  * Kp[1] * sizeof(h16));
    float* rbuf  = (float*)carve((size_t)BNH * sizeof(float));
    float* ubuf  = (float*)carve((size_t)BNH * sizeof(float));
    float* hbuf  = (float*)carve((size_t)BNH * sizeof(float));
    float* outs0 = (float*)carve((size_t)TSTEPS * BNH * sizeof(float));

    // zero A-matrix once: layer-0 pad columns (330..351) must be 0
    hipMemsetAsync(X, 0, (size_t)BN * 640 * sizeof(h16), stream);

    // one-time f16 weight repack into WMMA B-fragment order
    for (int l = 0; l < 2; ++l) {
        int Kdim = (l == 0) ? 330 : 640;
        int KT   = Kp[l] / 32;
        int tg = 128 * KT * 32;
        pack_weights<<<(tg + 255) / 256, 256, 0, stream>>>(Wg[l], Kdim, 128, KT, Wgp[l]);
        int tc = 64 * KT * 32;
        pack_weights<<<(tc + 255) / 256, 256, 0, stream>>>(Wc[l], Kdim, 64, KT, Wcp[l]);
    }

    for (int l = 0; l < 2; ++l) {
        const int in_f = (l == 0) ? 2 : HDIM;
        const int F    = in_f + HDIM;             // 66 / 128
        const int FB   = F * BATCH;               // 528 / 1024
        const int KT   = Kp[l] / 32;              // 11 / 20
        const int nEl  = N_NODES * FB;
        const int ewGrid = (nEl + 255) / 256;
        const size_t accBytes = (size_t)nEl * sizeof(float);

        hipMemcpyAsync(hbuf, init_h + (size_t)l * BNH, (size_t)BNH * sizeof(float),
                       hipMemcpyDeviceToDevice, stream);

        for (int t = 0; t < TSTEPS; ++t) {
            const float* xt = (l == 0)
                ? inputs + (size_t)t * BATCH * N_NODES * 2
                : outs0  + (size_t)t * BNH;

            // ---- pass 0: gates input diffusion; pass 1: candidate ----------
            for (int pass = 0; pass < 2; ++pass) {
                concat_build<<<ewGrid, 256, 0, stream>>>(
                    xt, in_f, hbuf, rbuf, pass, xcat, X, Kp[l]);

                for (int s = 0; s < 2; ++s) {
                    const int*   rs = srows + s * NNZE;
                    const int*   cs = scols + s * NNZE;
                    const float* vs = svals + s * NNZE;
                    // x1 = A x0
                    hipMemsetAsync(acc1, 0, accBytes, stream);
                    spmm_atomic<<<NNZE, 256, 0, stream>>>(rs, cs, vs, xcat, acc1, FB);
                    cheb_writeX<<<ewGrid, 256, 0, stream>>>(
                        acc1, acc1, 1.0f, 0.0f, X, Kp[l], F, 1 + 2 * s);
                    // x2 = 2 A x1 - x0
                    hipMemsetAsync(acc2, 0, accBytes, stream);
                    spmm_atomic<<<NNZE, 256, 0, stream>>>(rs, cs, vs, acc1, acc2, FB);
                    cheb_writeX<<<ewGrid, 256, 0, stream>>>(
                        acc2, xcat, 2.0f, -1.0f, X, Kp[l], F, 2 + 2 * s);
                }

                if (pass == 0) {
                    wmma_gates<<<BN / ROWS_PB, 256, 0, stream>>>(
                        X, Kp[l], KT, Wgp[l], bg[l], rbuf, ubuf);
                } else {
                    float* outs_t = ((l == 0) ? outs0 : cur_out) + (size_t)t * BNH;
                    wmma_cand_update<<<BN / ROWS_PB, 128, 0, stream>>>(
                        X, Kp[l], KT, Wcp[l], bc[l], ubuf, hbuf,
                        outs_t, hid_out + (size_t)l * BNH,
                        (t == TSTEPS - 1) ? 1 : 0);
                }
            }
        }
    }
}
